// Model_RNN_46969762349555
// MI455X (gfx1250) — compile-verified, matrix operand order checked by reading, not run on hardware
//
#include <hip/hip_runtime.h>
#include <math.h>

#define B_  64
#define T_  512
#define F_  768
#define E_  100
#define I_  868
#define KP_ 896    // K padded to multiple of 32
#define H_  64
#define NCH (KP_ / 32)          // 28 K-chunks
#define WPK_DWORDS (8 * NCH * 32 * 8)   // 8 ntiles x 28 chunks x 32 lanes x 8 dwords

typedef __attribute__((ext_vector_type(16))) __bf16 v16bf;
typedef __attribute__((ext_vector_type(8)))  float  v8f;

union Frag {
  v16bf    v;
  unsigned u[8];
  uint4    q[2];
};

// round-to-nearest-even f32 -> bf16 pair packed into one dword
__device__ __forceinline__ unsigned pack_bf16(float a, float b) {
  union { float f; unsigned u; } x, y;
  x.f = a; y.f = b;
  unsigned ra = (x.u + 0x7FFFu + ((x.u >> 16) & 1u)) >> 16;
  unsigned rb = (y.u + 0x7FFFu + ((y.u >> 16) & 1u)) >> 16;
  return (ra & 0xFFFFu) | (rb << 16);
}

// 16-bit A/B fragment layout (ISA 7.12.2): VGPR v, lane-half -> starting K of pair
__device__ __forceinline__ int kofs(int v, int half) {
  return ((v & 3) << 1) + ((v >> 2) << 4) + (half << 3);
}

// ---------------------------------------------------------------------------
// Kernel 0: one-shot pre-pack of W_ih_{f,b} into B-fragment bf16 layout.
// Wpk[((nt*28 + c)*32 + lane)*8 + v] = pack(W[h][k], W[h][k+1]) zero-padded.
// ---------------------------------------------------------------------------
__global__ __launch_bounds__(256)
void pack_w_kernel(const float* __restrict__ Wf, const float* __restrict__ Wb,
                   unsigned* __restrict__ Wpk) {
  const int id = blockIdx.x * 256 + threadIdx.x;
  if (id >= WPK_DWORDS) return;
  const int v    = id & 7;
  const int lane = (id >> 3) & 31;
  const int c    = (id >> 8) % NCH;
  const int nt   = id / (NCH * 256);
  const int dir = nt >> 2, ntl = nt & 3;
  const int n = lane & 15, half = lane >> 4;
  const int h = ntl * 16 + n;
  const int k = c * 32 + kofs(v, half);
  const float* W = dir ? Wb : Wf;
  float w0 = 0.f, w1 = 0.f;
  if (k < I_) { w0 = W[h * I_ + k]; w1 = W[h * I_ + k + 1]; }
  Wpk[id] = pack_bf16(w0, w1);
}

// ---------------------------------------------------------------------------
// Kernel 1: fused gather/concat + input projection GEMM (bf16 WMMA, f32 acc)
// 32 rows per block (2 M-tiles), 8 waves = 4 N-tiles x 2 directions.
// Inner loop: 4x ds_load_b128 (A) + 2x global b128 (B) + 2x WMMA.
// ---------------------------------------------------------------------------
__global__ __launch_bounds__(256)
void rnn_proj_kernel(const float* __restrict__ inputs, const int* __restrict__ pos,
                     const float* __restrict__ emb,
                     const unsigned* __restrict__ Wpk,
                     const float* __restrict__ bihf, const float* __restrict__ bhhf,
                     const float* __restrict__ bihb, const float* __restrict__ bhhb,
                     float* __restrict__ xp) {
  __shared__ unsigned ldsA[32 * (KP_ / 2)];  // 32 rows x 448 packed bf16 pairs (56KB)
  const int tid = threadIdx.x;
  const int mt  = blockIdx.x;                // 32-row tile index

  // cooperative stage of the 32 x 896 A-tile (concat + zero pad), bf16-packed
  for (int i = tid; i < 32 * (KP_ / 2); i += 256) {
    const int m = i / (KP_ / 2);
    const int p = i % (KP_ / 2);
    const int k = p * 2;
    const int r = mt * 32 + m;       // r = b*T + t (tiles never straddle b: 32 | T)
    const int b = r / T_, t = r % T_;
    float f0 = 0.f, f1 = 0.f;
    if (k < F_) {
      const float* base = inputs + ((size_t)b * T_ + t) * F_;
      f0 = base[k]; f1 = base[k + 1];
    } else if (k < I_) {
      const int    pv = pos[b * T_ + t];
      const float* er = emb + (size_t)pv * E_;
      const int    kk = k - F_;
      f0 = er[kk]; f1 = er[kk + 1];
    }
    ldsA[i] = pack_bf16(f0, f1);
  }
  __syncthreads();

  const int w = tid >> 5, lane = tid & 31;
  const int dir = w >> 2, ntl = w & 3;
  const int n = lane & 15, half = lane >> 4;
  const int h = ntl * 16 + n;
  const int arow = lane & 15;

  v8f acc0 = {}, acc1 = {};
  for (int c = 0; c < NCH; ++c) {
    Frag a0, a1, bf;
    const int abase = c * 16 + half * 4;     // dword index of q[0] within a row
    a0.q[0] = *(const uint4*)&ldsA[arow * (KP_ / 2) + abase];
    a0.q[1] = *(const uint4*)&ldsA[arow * (KP_ / 2) + abase + 8];
    a1.q[0] = *(const uint4*)&ldsA[(arow + 16) * (KP_ / 2) + abase];
    a1.q[1] = *(const uint4*)&ldsA[(arow + 16) * (KP_ / 2) + abase + 8];
    const unsigned gb = (((unsigned)w * NCH + c) * 32 + lane) * 8;
    bf.q[0] = *(const uint4*)&Wpk[gb];
    bf.q[1] = *(const uint4*)&Wpk[gb + 4];
    acc0 = __builtin_amdgcn_wmma_f32_16x16x32_bf16(false, a0.v, false, bf.v,
                                                   (short)0, acc0, false, false);
    acc1 = __builtin_amdgcn_wmma_f32_16x16x32_bf16(false, a1.v, false, bf.v,
                                                   (short)0, acc1, false, false);
  }

  const float bias = dir ? (bihb[h] + bhhb[h]) : (bihf[h] + bhhf[h]);
#pragma unroll
  for (int r = 0; r < 8; ++r) {
    const int m  = r + half * 8;             // C/D layout: lanes>=16 hold M+8
    {
      const int rg = mt * 32 + m;
      const int b  = rg / T_, t = rg % T_;
      xp[(((size_t)dir * T_ + t) * B_ + b) * H_ + h] = acc0[r] + bias;
    }
    {
      const int rg = mt * 32 + 16 + m;
      const int b  = rg / T_, t = rg % T_;
      xp[(((size_t)dir * T_ + t) * B_ + b) * H_ + h] = acc1[r] + bias;
    }
  }
}

// ---------------------------------------------------------------------------
// Kernel 2: recurrent scan. grid=2 (dir), block=512 (16 waves = 4x4 MN tiles).
// h_t = tanh(xp_t + h_{t-1} @ W_hh^T), h kept in LDS as bf16 between steps.
// ---------------------------------------------------------------------------
__global__ __launch_bounds__(512)
void rnn_scan_kernel(const float* __restrict__ xp,
                     const float* __restrict__ Whhf, const float* __restrict__ Whhb,
                     float* __restrict__ hs) {
  __shared__ unsigned hlds[B_ * (H_ / 2)];   // 64 rows x 32 packed bf16 pairs (8KB)
  const int tid = threadIdx.x;
  const int dir = blockIdx.x;
  for (int i = tid; i < B_ * (H_ / 2); i += 512) hlds[i] = 0;

  const int w = tid >> 5, lane = tid & 31;
  const int mtile = w >> 2, ntile = w & 3;
  const int n = lane & 15, half = lane >> 4;
  const int hcol = ntile * 16 + n;           // output feature column
  const float* Whh = dir ? Whhb : Whhf;

  // B[k][n] = W_hh[n][k]; constant across steps -> preload both K-chunks
  Frag bf0, bf1;
#pragma unroll
  for (int v = 0; v < 8; ++v) {
    const int k0 = kofs(v, half);
    bf0.u[v] = pack_bf16(Whh[hcol * H_ + k0],      Whh[hcol * H_ + k0 + 1]);
    bf1.u[v] = pack_bf16(Whh[hcol * H_ + 32 + k0], Whh[hcol * H_ + 32 + k0 + 1]);
  }
  __syncthreads();

  const int mrow = mtile * 16 + (lane & 15); // batch row for A fragment
  const int abase = mrow * (H_ / 2) + half * 4;
  for (int tt = 0; tt < T_; ++tt) {
    const int t = dir ? (T_ - 1 - tt) : tt;

    Frag a0, a1;                             // A = h (bf16) from LDS, b128 loads
    a0.q[0] = *(const uint4*)&hlds[abase];
    a0.q[1] = *(const uint4*)&hlds[abase + 8];
    a1.q[0] = *(const uint4*)&hlds[abase + 16];
    a1.q[1] = *(const uint4*)&hlds[abase + 24];

    const size_t xbase = (((size_t)dir * T_ + t) * B_) * H_;
    v8f c;
#pragma unroll
    for (int r = 0; r < 8; ++r) {
      const int b = mtile * 16 + r + half * 8;
      c[r] = xp[xbase + (size_t)b * H_ + hcol];
    }

    __syncthreads();                         // all A reads complete (WAR)

    c = __builtin_amdgcn_wmma_f32_16x16x32_bf16(false, a0.v, false, bf0.v,
                                                (short)0, c, false, false);
    c = __builtin_amdgcn_wmma_f32_16x16x32_bf16(false, a1.v, false, bf1.v,
                                                (short)0, c, false, false);

    unsigned short* hl16 = (unsigned short*)hlds;
#pragma unroll
    for (int r = 0; r < 8; ++r) {
      const int   b  = mtile * 16 + r + half * 8;
      const float hv = tanhf(c[r]);
      hs[xbase + (size_t)b * H_ + hcol] = hv;  // f32 output for head
      union { float f; unsigned u; } x; x.f = hv;
      hl16[b * H_ + hcol] =
          (unsigned short)((x.u + 0x7FFFu + ((x.u >> 16) & 1u)) >> 16);
    }
    __syncthreads();                         // new h visible (RAW)
  }
}

// ---------------------------------------------------------------------------
// Kernel 3: head. Linear(128,32) -> LayerNorm(32) -> ReLU -> Linear(32,4)
// ---------------------------------------------------------------------------
__global__ __launch_bounds__(256)
void head_kernel(const float* __restrict__ hs,
                 const float* __restrict__ W1, const float* __restrict__ b1,
                 const float* __restrict__ gamma, const float* __restrict__ beta,
                 const float* __restrict__ W2, const float* __restrict__ b2,
                 float* __restrict__ out) {
  __shared__ float W1s[32 * 128];
  const int tid = threadIdx.x;
  for (int i = tid; i < 32 * 128; i += 256) W1s[i] = W1[i];
  __syncthreads();

  const int idx = blockIdx.x * 256 + tid;    // idx = b*T + t
  const int b = idx >> 9, t = idx & (T_ - 1);

  float h1[32];
#pragma unroll
  for (int o = 0; o < 32; ++o) h1[o] = b1[o];

  const float* hf = hs + (((size_t)0 * T_ + t) * B_ + b) * H_;
  const float* hb = hs + (((size_t)1 * T_ + t) * B_ + b) * H_;
  for (int j = 0; j < H_; ++j) {
    const float x = hf[j];
#pragma unroll
    for (int o = 0; o < 32; ++o) h1[o] = fmaf(x, W1s[o * 128 + j], h1[o]);
  }
  for (int j = 0; j < H_; ++j) {
    const float x = hb[j];
#pragma unroll
    for (int o = 0; o < 32; ++o) h1[o] = fmaf(x, W1s[o * 128 + 64 + j], h1[o]);
  }

  float mu = 0.f;
#pragma unroll
  for (int o = 0; o < 32; ++o) mu += h1[o];
  mu *= (1.0f / 32.0f);
  float var = 0.f;
#pragma unroll
  for (int o = 0; o < 32; ++o) { const float d = h1[o] - mu; var = fmaf(d, d, var); }
  var *= (1.0f / 32.0f);
  const float rs = rsqrtf(var + 1e-5f);
#pragma unroll
  for (int o = 0; o < 32; ++o) {
    float v = (h1[o] - mu) * rs * gamma[o] + beta[o];
    h1[o] = v > 0.f ? v : 0.f;               // ReLU (dropout = identity)
  }

#pragma unroll
  for (int o2 = 0; o2 < 4; ++o2) {
    float s = b2[o2];
#pragma unroll
    for (int o = 0; o < 32; ++o) s = fmaf(h1[o], W2[o2 * 32 + o], s);
    out[(size_t)idx * 4 + o2] = s;
  }
}

// ---------------------------------------------------------------------------
extern "C" void kernel_launch(void* const* d_in, const int* in_sizes, int n_in,
                              void* d_out, int out_size, void* d_ws, size_t ws_size,
                              hipStream_t stream) {
  (void)in_sizes; (void)n_in; (void)out_size; (void)ws_size;
  const float* inputs = (const float*)d_in[0];
  const int*   pos    = (const int*)d_in[1];
  const float* emb    = (const float*)d_in[2];
  const float* W_ih_f = (const float*)d_in[3];
  const float* W_hh_f = (const float*)d_in[4];
  const float* b_ih_f = (const float*)d_in[5];
  const float* b_hh_f = (const float*)d_in[6];
  const float* W_ih_b = (const float*)d_in[7];
  const float* W_hh_b = (const float*)d_in[8];
  const float* b_ih_b = (const float*)d_in[9];
  const float* b_hh_b = (const float*)d_in[10];
  const float* W1     = (const float*)d_in[11];
  const float* b1     = (const float*)d_in[12];
  const float* gamma  = (const float*)d_in[13];
  const float* beta   = (const float*)d_in[14];
  const float* W2     = (const float*)d_in[15];
  const float* b2     = (const float*)d_in[16];

  unsigned* Wpk = (unsigned*)d_ws;                    // 229 KB packed weights
  float*    xp  = (float*)d_ws + WPK_DWORDS;          // [2][T][B][H]  16 MB
  float*    hsb = xp + (size_t)2 * T_ * B_ * H_;      // [2][T][B][H]  16 MB

  pack_w_kernel<<<(WPK_DWORDS + 255) / 256, 256, 0, stream>>>(W_ih_f, W_ih_b, Wpk);
  rnn_proj_kernel<<<(B_ * T_) / 32, 256, 0, stream>>>(
      inputs, pos, emb, Wpk, b_ih_f, b_hh_f, b_ih_b, b_hh_b, xp);
  rnn_scan_kernel<<<2, 512, 0, stream>>>(xp, W_hh_f, W_hh_b, hsb);
  head_kernel<<<(B_ * T_) / 256, 256, 0, stream>>>(
      hsb, W1, b1, gamma, beta, W2, b2, (float*)d_out);
}
